// SparseVLMModel_90117003805189
// MI455X (gfx1250) — compile-verified, matrix operand order checked by reading, not run on hardware
//
#include <hip/hip_runtime.h>
#include <hip/hip_bf16.h>

// Problem constants (match reference setup_inputs)
constexpr int B  = 32;
constexpr int LV = 576;
constexpr int LT = 512;
constexpr int DD = 1024;
constexpr int DT = 4096;
constexpr float LAMBDA = 0.5f;

typedef __attribute__((ext_vector_type(16))) __bf16 v16bf;
typedef __attribute__((ext_vector_type(8)))  float  v8f;
typedef __attribute__((ext_vector_type(4)))  unsigned int v4u;

union Frag { v16bf h; v4u u[2]; };

__device__ __forceinline__ unsigned short f32_to_bf16(float f) {
  unsigned int u = __float_as_uint(f);
  unsigned int r = u + 0x7FFFu + ((u >> 16) & 1u);   // round-to-nearest-even
  return (unsigned short)(r >> 16);
}

// Generic LDS pointer -> 32-bit LDS byte offset (low 32 bits of the address).
__device__ __forceinline__ unsigned lds_off32(const void* p) {
  return (unsigned)(unsigned long long)p;
}

// CDNA5 async copy engine, GVS mode: mem_addr = SGPR base + VGPR offset.
// 16B per lane, global -> LDS, tracked by ASYNCcnt. SALU-only K advance.
__device__ __forceinline__ void async_b128_s(unsigned lds, unsigned voff,
                                             unsigned long long base) {
  asm volatile("global_load_async_to_lds_b128 %0, %1, %2"
               :: "v"(lds), "v"(voff), "s"(base) : "memory");
}
__device__ __forceinline__ void wait_async6() {
  asm volatile("s_wait_asynccnt 0x6" ::: "memory");
}
__device__ __forceinline__ void wait_async0() {
  asm volatile("s_wait_asynccnt 0x0" ::: "memory");
}

// ---------------------------------------------------------------------------
// f32 -> bf16 conversion, vectorized (b128 in, b64 out). n4 = n/4.
// ---------------------------------------------------------------------------
__global__ void cvt_bf16_kernel(const float4* __restrict__ src,
                                uint2* __restrict__ dst, size_t n4) {
  size_t i = (size_t)blockIdx.x * blockDim.x + threadIdx.x;
  size_t stride = (size_t)gridDim.x * blockDim.x;
  for (; i < n4; i += stride) {
    float4 f = src[i];
    unsigned int lo = ((unsigned int)f32_to_bf16(f.y) << 16) | f32_to_bf16(f.x);
    unsigned int hi = ((unsigned int)f32_to_bf16(f.w) << 16) | f32_to_bf16(f.z);
    dst[i] = make_uint2(lo, hi);
  }
}

// ---------------------------------------------------------------------------
// Batched WMMA GEMM: C[m,n] = sum_k A[m,k] * Bm[n,k]   (A,B bf16; C f32)
// Block tile 64x128, 128 threads = 4 waves, each wave owns a 32x64 sub-tile
// (2x4 v_wmma_f32_16x16x32_bf16 accumulators). K stepped by 32.
// Global->LDS staging uses global_load_async_to_lds_b128 (GVS: uniform SGPR
// base + invariant per-thread VGPR offset) with double-buffered LDS: tile k+1
// streams in (6 async ops/thread) while tile k feeds the WMMAs;
// s_wait_asynccnt 6 retires exactly the previous tile (in-order completion).
// Last tile is peeled so the steady-state loop is branch-free.
// PROJ epilogue: +bias[n], zero rows >= keep[b].
// ---------------------------------------------------------------------------
template<bool PROJ>
__launch_bounds__(128)
__global__ void wmma_gemm_kernel(const unsigned short* __restrict__ Ab,
                                 const unsigned short* __restrict__ Bb,
                                 float* __restrict__ Cb,
                                 const float* __restrict__ bias,
                                 const int* __restrict__ keep,
                                 int M, int N, int K,
                                 long long aStride, long long bStride,
                                 long long cStride) {
  const int b = blockIdx.z;
  const unsigned short* __restrict__ A  = Ab + (size_t)b * aStride;
  const unsigned short* __restrict__ Bm = Bb + (size_t)b * bStride;
  float* __restrict__ C = Cb + (size_t)b * cStride;

  const int m0 = blockIdx.y * 64;
  const int n0 = blockIdx.x * 128;

  __shared__ __align__(16) unsigned short As[2][64 * 32];
  __shared__ __align__(16) unsigned short Bs[2][128 * 32];

  const int t    = threadIdx.x;
  const int lane = t & 31;
  const int wave = t >> 5;
  const int lid  = lane & 15;         // row/col within 16x16 tile
  const int g    = lane >> 4;         // lane-group selects K halves / M+8 rows
  const int mo   = (wave & 1) * 32;
  const int no   = (wave >> 1) * 64;

  // Per-thread copy slots: A tile = 256 16B chunks (2/thread),
  // B tile = 512 16B chunks (4/thread). chunk c -> row c>>2, col (c&3)*8.
  // Loop-invariant: LDS dest offsets and per-thread global byte offsets.
  unsigned vA[2], vB[4];
  unsigned lA[2][2], lB[2][4];
  #pragma unroll
  for (int q = 0; q < 2; ++q) {
    const int c = t + q * 128, r = c >> 2, cc = (c & 3) << 3;
    vA[q] = (unsigned)(((m0 + r) * K + cc) * 2);
    lA[0][q] = lds_off32(&As[0][r * 32 + cc]);
    lA[1][q] = lds_off32(&As[1][r * 32 + cc]);
  }
  #pragma unroll
  for (int q = 0; q < 4; ++q) {
    const int c = t + q * 128, r = c >> 2, cc = (c & 3) << 3;
    vB[q] = (unsigned)(((n0 + r) * K + cc) * 2);
    lB[0][q] = lds_off32(&Bs[0][r * 32 + cc]);
    lB[1][q] = lds_off32(&Bs[1][r * 32 + cc]);
  }

  // Uniform bases advanced by one K-tile (64B) per issue -- pure SALU.
  unsigned long long baseA = (unsigned long long)A;
  unsigned long long baseB = (unsigned long long)Bm;

  v8f acc[2][4] = {};

  auto compute_tile = [&](int cb) {
    // Fragment loads per the CDNA5 16-bit A/B layout:
    // lane<16 : elems 0..7 -> K 0..7,  elems 8..15 -> K 16..23
    // lane>=16: elems 0..7 -> K 8..15, elems 8..15 -> K 24..31
    // => two contiguous 16B ds_load_b128 per fragment at offsets 8g, 16+8g.
    Frag fa[2], fb[4];
    #pragma unroll
    for (int i = 0; i < 2; ++i) {
      const int m = mo + 16 * i + lid;
      fa[i].u[0] = *(const v4u*)&As[cb][m * 32 + 8 * g];
      fa[i].u[1] = *(const v4u*)&As[cb][m * 32 + 16 + 8 * g];
    }
    #pragma unroll
    for (int j = 0; j < 4; ++j) {
      const int n = no + 16 * j + lid;
      fb[j].u[0] = *(const v4u*)&Bs[cb][n * 32 + 8 * g];
      fb[j].u[1] = *(const v4u*)&Bs[cb][n * 32 + 16 + 8 * g];
    }
    #pragma unroll
    for (int i = 0; i < 2; ++i)
      #pragma unroll
      for (int j = 0; j < 4; ++j)
        acc[i][j] = __builtin_amdgcn_wmma_f32_16x16x32_bf16(
            false, fa[i].h, false, fb[j].h, (short)0, acc[i][j], false, false);
  };

  // Prologue: stream tile 0 into buffer 0.
  #pragma unroll
  for (int q = 0; q < 2; ++q) async_b128_s(lA[0][q], vA[q], baseA);
  #pragma unroll
  for (int q = 0; q < 4; ++q) async_b128_s(lB[0][q], vB[q], baseB);
  baseA += 64; baseB += 64;

  // Steady state: branch-free; always prefetch the next tile.
  int cur = 0;
  for (int k0 = 0; k0 < K - 32; k0 += 32) {
    const int nxt = cur ^ 1;
    #pragma unroll
    for (int q = 0; q < 2; ++q) async_b128_s(lA[nxt][q], vA[q], baseA);
    #pragma unroll
    for (int q = 0; q < 4; ++q) async_b128_s(lB[nxt][q], vB[q], baseB);
    baseA += 64; baseB += 64;

    wait_async6();        // previous tile's 6 copies have landed (in-order)
    __syncthreads();
    compute_tile(cur);
    __syncthreads();      // all reads of buf(cur) done before it refills
    cur = nxt;
  }

  // Peeled last tile.
  wait_async0();
  __syncthreads();
  compute_tile(cur);

  // Epilogue. C/D layout: VGPR r -> row (r + 8g), lanes 0..15 -> col lid.
  const int kb = PROJ ? keep[b] : 0;
  #pragma unroll
  for (int i = 0; i < 2; ++i) {
    #pragma unroll
    for (int j = 0; j < 4; ++j) {
      const int col = n0 + no + 16 * j + lid;
      #pragma unroll
      for (int r = 0; r < 8; ++r) {
        const int row = m0 + mo + 16 * i + 8 * g + r;
        float v = acc[i][j][r];
        if (PROJ) {
          v += bias[col];
          v = (row < kb) ? v : 0.0f;
        }
        C[(size_t)row * N + col] = v;
      }
    }
  }
}

// ---------------------------------------------------------------------------
// Row softmax over LT=512 (one block of 256 threads per (b,v) row, in place).
// ---------------------------------------------------------------------------
__global__ void softmax_kernel(float* __restrict__ attn) {
  const int row = blockIdx.x;
  float* p = attn + (size_t)row * LT;
  const int t = threadIdx.x;
  float x0 = p[t], x1 = p[t + 256];

  __shared__ float red[256];
  red[t] = fmaxf(x0, x1);
  __syncthreads();
  for (int s = 128; s > 0; s >>= 1) {
    if (t < s) red[t] = fmaxf(red[t], red[t + s]);
    __syncthreads();
  }
  const float mx = red[0];
  __syncthreads();

  const float e0 = __expf(x0 - mx), e1 = __expf(x1 - mx);
  red[t] = e0 + e1;
  __syncthreads();
  for (int s = 128; s > 0; s >>= 1) {
    if (t < s) red[t] += red[t + s];
    __syncthreads();
  }
  const float inv = 1.0f / red[0];
  p[t] = e0 * inv;
  p[t + 256] = e1 * inv;
}

// ---------------------------------------------------------------------------
// rel[b,t] = mean_v attn ; raters = rel >= mean_t rel ; denom ; rank surrogate
// rank(attn_m^T) == #nonzero rows == #raters (generic position) -> keep[b].
// One block (512 threads) per batch.
// ---------------------------------------------------------------------------
__global__ void rel_raters_kernel(const float* __restrict__ attn,
                                  float* __restrict__ raters,
                                  float* __restrict__ denom,
                                  int* __restrict__ keep) {
  const int b = blockIdx.x, t = threadIdx.x;
  const float* pb = attn + (size_t)b * LV * LT;
  float s = 0.0f;
  for (int v = 0; v < LV; ++v) s += pb[(size_t)v * LT + t];
  const float rel = s * (1.0f / LV);

  __shared__ float red[512];
  red[t] = rel;
  __syncthreads();
  for (int st = 256; st > 0; st >>= 1) {
    if (t < st) red[t] += red[t + st];
    __syncthreads();
  }
  const float mean = red[0] * (1.0f / LT);
  __syncthreads();

  const float r = (rel >= mean) ? 1.0f : 0.0f;
  raters[b * LT + t] = r;
  red[t] = r;
  __syncthreads();
  for (int st = 256; st > 0; st >>= 1) {
    if (t < st) red[t] += red[t + st];
    __syncthreads();
  }
  if (t == 0) {
    const float nr = red[0];
    denom[b] = fmaxf(nr, 1e-5f);
    int rank = (int)nr;
    if (rank > LV) rank = LV;
    const int prune = (int)(LAMBDA * (float)(LV - rank));
    keep[b] = LV - prune;
  }
}

// ---------------------------------------------------------------------------
// vision_scores[b,v] = sum_t attn*raters / denom[b]. Block per (b,v) row.
// ---------------------------------------------------------------------------
__global__ void scores_kernel(const float* __restrict__ attn,
                              const float* __restrict__ raters,
                              const float* __restrict__ denom,
                              float* __restrict__ scores) {
  const int row = blockIdx.x;
  const int b = row / LV;
  const float* p = attn + (size_t)row * LT;
  const float* r = raters + b * LT;
  const int t = threadIdx.x;
  __shared__ float red[256];
  red[t] = p[t] * r[t] + p[t + 256] * r[t + 256];
  __syncthreads();
  for (int s = 128; s > 0; s >>= 1) {
    if (t < s) red[t] += red[t + s];
    __syncthreads();
  }
  if (t == 0) scores[row] = red[0] / denom[b];
}

// ---------------------------------------------------------------------------
// Descending rank-sort of 576 scores per batch (tie-break: lower index first,
// matching stable argsort of -scores). order[b][rank] = token index.
// ---------------------------------------------------------------------------
__global__ void order_kernel(const float* __restrict__ scores,
                             int* __restrict__ order) {
  const int b = blockIdx.x, t = threadIdx.x;
  __shared__ float s[LV];
  s[t] = scores[b * LV + t];
  __syncthreads();
  const float mine = s[t];
  int rank = 0;
  for (int u = 0; u < LV; ++u) {
    const float su = s[u];
    rank += (su > mine) || (su == mine && u < t);
  }
  order[b * LV + rank] = t;
}

// ---------------------------------------------------------------------------
// Gather sorted bf16 token rows: sorted[b][p][:] = patch_bf16[b][order[b][p]][:]
// ---------------------------------------------------------------------------
__global__ void gather_kernel(const unsigned short* __restrict__ patch_bf,
                              const int* __restrict__ order,
                              unsigned short* __restrict__ sorted_bf) {
  const int p = blockIdx.x, b = blockIdx.y;
  const int src = order[b * LV + p];
  const uint2* s = (const uint2*)patch_bf + ((size_t)b * LV + src) * (DD / 4);
  uint2* d = (uint2*)sorted_bf + ((size_t)b * LV + p) * (DD / 4);
  for (int i = threadIdx.x; i < DD / 4; i += 256) d[i] = s[i];
}

// ---------------------------------------------------------------------------
// Second output: valid mask (1.0 / 0.0) appended after visual.
// ---------------------------------------------------------------------------
__global__ void valid_kernel(float* __restrict__ out,
                             const int* __restrict__ keep, int out_size) {
  const int i = blockIdx.x * blockDim.x + threadIdx.x;
  const int n = B * LV;
  const long long base = (long long)B * LV * DT;
  if (i < n && (long long)out_size >= base + n) {
    const int b = i / LV, v = i % LV;
    out[base + i] = (v < keep[b]) ? 1.0f : 0.0f;
  }
}

extern "C" void kernel_launch(void* const* d_in, const int* in_sizes, int n_in,
                              void* d_out, int out_size, void* d_ws, size_t ws_size,
                              hipStream_t stream) {
  const float* patch = (const float*)d_in[0];
  const float* text  = (const float*)d_in[1];
  const float* projw = (const float*)d_in[2];
  const float* projb = (const float*)d_in[3];
  float* out = (float*)d_out;

  char* ws = (char*)d_ws;
  size_t off = 0;
  auto carve = [&](size_t bytes) -> void* {
    void* p = ws + off;
    off += (bytes + 255) & ~(size_t)255;
    return p;
  };

  unsigned short* patch_bf  = (unsigned short*)carve((size_t)B * LV * DD * 2);
  unsigned short* text_bf   = (unsigned short*)carve((size_t)B * LT * DD * 2);
  unsigned short* projw_bf  = (unsigned short*)carve((size_t)DT * DD * 2);
  unsigned short* sorted_bf = (unsigned short*)carve((size_t)B * LV * DD * 2);
  float* attn   = (float*)carve((size_t)B * LV * LT * 4);
  float* raters = (float*)carve((size_t)B * LT * 4);
  float* denom  = (float*)carve((size_t)B * 4);
  int*   keep   = (int*)carve((size_t)B * 4);
  float* scores = (float*)carve((size_t)B * LV * 4);
  int*   order  = (int*)carve((size_t)B * LV * 4);

  // 1) bf16 conversions
  {
    size_t n4;
    n4 = (size_t)B * LV * DD / 4;
    cvt_bf16_kernel<<<(unsigned)((n4 + 255) / 256), 256, 0, stream>>>(
        (const float4*)patch, (uint2*)patch_bf, n4);
    n4 = (size_t)B * LT * DD / 4;
    cvt_bf16_kernel<<<(unsigned)((n4 + 255) / 256), 256, 0, stream>>>(
        (const float4*)text, (uint2*)text_bf, n4);
    n4 = (size_t)DT * DD / 4;
    cvt_bf16_kernel<<<(unsigned)((n4 + 255) / 256), 256, 0, stream>>>(
        (const float4*)projw, (uint2*)projw_bf, n4);
  }

  // 2) S = P * T^T  (per batch), f32 scores into attn buffer
  {
    dim3 grid(LT / 128, LV / 64, B);
    wmma_gemm_kernel<false><<<grid, 128, 0, stream>>>(
        patch_bf, text_bf, attn, nullptr, nullptr,
        LV, LT, DD,
        (long long)LV * DD, (long long)LT * DD, (long long)LV * LT);
  }

  // 3) softmax over text tokens
  softmax_kernel<<<B * LV, 256, 0, stream>>>(attn);

  // 4) relevance, raters, rank surrogate -> keep
  rel_raters_kernel<<<B, 512, 0, stream>>>(attn, raters, denom, keep);

  // 5) vision scores
  scores_kernel<<<B * LV, 256, 0, stream>>>(attn, raters, denom, scores);

  // 6) descending order per batch
  order_kernel<<<B, LV, 0, stream>>>(scores, order);

  // 7) gather sorted bf16 tokens
  {
    dim3 grid(LV, B);
    gather_kernel<<<grid, 256, 0, stream>>>(patch_bf, order, sorted_bf);
  }

  // 8) visual = sorted * W^T + b, masked by valid  -> d_out
  {
    dim3 grid(DT / 128, LV / 64, B);
    wmma_gemm_kernel<true><<<grid, 128, 0, stream>>>(
        sorted_bf, projw_bf, out, projb, keep,
        LV, DT, DD,
        (long long)LV * DD, 0LL, (long long)LV * DT);
  }

  // 9) valid mask tail
  valid_kernel<<<(B * LV + 255) / 256, 256, 0, stream>>>(out, keep, out_size);
}
